// CrAKN_87849261072555
// MI455X (gfx1250) — matrix operand order repaired and checked by reading, not compiled, not run on hardware
//
#include <hip/hip_runtime.h>
#include <cstdint>
#include <cstddef>

// ---------------------------------------------------------------------------
// CrAKN forward for MI455X (gfx1250, wave32, WMMA).
// Dominant work: per-layer bias MLP 128->512->128 over 65536 rows, fused into
// one WMMA kernel with the 512-wide intermediate staged in (dynamic) LDS.
// 64 rows per workgroup: each bpW/bowW B-fragment is reused by 4 independent
// WMMA accumulator chains (hides the f16 WMMA->WMMA hazard, cuts L2 weight
// traffic 4x). Accumulators zero-init so WMMA uses inline SRC2=0 (bias added
// in the epilogue). Edge duplication halves all big GEMMs; the [1-I | I] mask
// is folded analytically into the softmax.
// ---------------------------------------------------------------------------

typedef _Float16 half_t;
typedef __attribute__((ext_vector_type(16))) _Float16 v16h;
typedef __attribute__((ext_vector_type(8)))  float    v8f;
typedef __attribute__((ext_vector_type(4)))  unsigned int u32x4;

#define NN    256     // crystals
#define FBF   256     // backbone features
#define DD    64      // embedding dim
#define HH    4       // heads
#define DHD   128     // head dim
#define LLAY  4       // layers
#define BDB   128     // bias dim
#define AKK   100     // edge feature dim
#define HDH   512     // HH*DHD
#define HBD   512     // HH*BDB
#define RROWS 65536   // NN*NN rows of the bias tensor
#define LDSLD 520     // LDS row stride (halves): 1040B, 16B aligned,
                      // 260 dwords % 64 banks = 4 -> conflict-free b128 reads
#define ROWS_PER_WG 64

// ---------------- WMMA fragment loaders (16x16x32 f16) ---------------------

// A (16x32, MxK), row-major source with row stride ld (halves).
// lane: M = lane&15 ; K in [k0+hi*16, +16) contiguous.
__device__ __forceinline__ v16h load_a_frag(const half_t* base, int row0, int ld,
                                            int k0, int lane) {
  int lh = lane & 15, hi = lane >> 4;
  const half_t* p = base + (size_t)(row0 + lh) * ld + k0 + hi * 16;
  union { u32x4 q[2]; v16h v; } u;
  u.q[0] = *(const u32x4*)(p);
  u.q[1] = *(const u32x4*)(p + 8);
  return u.v;
}

// B (32x16, KxN) from packed buffer: record (frag,lane) = 16 consecutive halves.
__device__ __forceinline__ v16h load_b_packed(const half_t* base, int frag, int lane) {
  const half_t* p = base + ((size_t)frag * 32 + lane) * 16;
  union { u32x4 q[2]; v16h v; } u;
  u.q[0] = *(const u32x4*)(p);
  u.q[1] = *(const u32x4*)(p + 8);
  return u.v;
}

// B (32x16) = M^T from row-major M [rows, ld]: B[K][Nc] = M[m0+Nc][k0+K].
__device__ __forceinline__ v16h load_bT_rows(const half_t* base, int m0, int ld,
                                             int k0, int lane) {
  int lh = lane & 15, hi = lane >> 4;
  const half_t* p = base + (size_t)(m0 + lh) * ld + k0 + hi * 16;
  union { u32x4 q[2]; v16h v; } u;
  u.q[0] = *(const u32x4*)(p);
  u.q[1] = *(const u32x4*)(p + 8);
  return u.v;
}

__device__ __forceinline__ float mishf(float x) {
  float sp = (x > 20.0f) ? x : log1pf(expf(x));
  return x * tanhf(sp);
}

__device__ __forceinline__ float wred_sum(float v) {
  #pragma unroll
  for (int o = 16; o > 0; o >>= 1) v += __shfl_xor(v, o, 32);
  return v;
}
__device__ __forceinline__ float wred_max(float v) {
  #pragma unroll
  for (int o = 16; o > 0; o >>= 1) v = fmaxf(v, __shfl_xor(v, o, 32));
  return v;
}

// ---------------- weight packing ------------------------------------------
// dst[((kk*nJ + j)*32 + lane)*16 + e] = src[kk*32 + (lane>>4)*16 + e][j*16 + (lane&15)]
__global__ void pack_b_kernel(const float* __restrict__ src, half_t* __restrict__ dst,
                              int K, int Nc) {
  int o = blockIdx.x * blockDim.x + threadIdx.x;
  if (o >= K * Nc) return;
  int nJ = Nc >> 4;
  int f = o >> 9, rem = o & 511;
  int lane = rem >> 4, e = rem & 15;
  int kk = f / nJ, j = f - kk * nJ;
  int Kr = kk * 32 + (lane >> 4) * 16 + e;
  int col = j * 16 + (lane & 15);
  dst[o] = (half_t)src[(size_t)Kr * Nc + col];
}

// Same, with source row permutation: effective weight row for k-index K
// (=h*128+dh in pq's natural layout) is W[dh*4 + h]. Requires K == 512.
__global__ void pack_perm_kernel(const float* __restrict__ src, half_t* __restrict__ dst,
                                 int K, int Nc) {
  int o = blockIdx.x * blockDim.x + threadIdx.x;
  if (o >= K * Nc) return;
  int nJ = Nc >> 4;
  int f = o >> 9, rem = o & 511;
  int lane = rem >> 4, e = rem & 15;
  int kk = f / nJ, j = f - kk * nJ;
  int Kr = kk * 32 + (lane >> 4) * 16 + e;
  int row = (Kr & 127) * 4 + (Kr >> 7);
  int col = j * 16 + (lane & 15);
  dst[o] = (half_t)src[(size_t)row * Nc + col];
}

__global__ void f32_to_f16_kernel(const float* __restrict__ s, half_t* __restrict__ d, int n) {
  int i = blockIdx.x * blockDim.x + threadIdx.x;
  if (i < n) d[i] = (half_t)s[i];
}

// ---------------- small precompute ----------------------------------------

__global__ void pred0_kernel(const float* __restrict__ X, const float* __restrict__ w,
                             const float* __restrict__ b, float* __restrict__ pred) {
  int n = threadIdx.x;
  float s = b[0];
  for (int a = 0; a < FBF; ++a) s += X[(size_t)n * FBF + a] * w[a];
  pred[n] = s;
}

// cdist + RBF expansion; one block per (n,m), 128 threads (one per center).
__global__ void rbf_kernel(const float* __restrict__ E, half_t* __restrict__ bias) {
  __shared__ float red[128];
  int n = blockIdx.x >> 8, m = blockIdx.x & 255;
  int t = threadIdx.x;
  float p = 0.0f;
  for (int a = t; a < AKK; a += 128) {
    float df = E[(size_t)n * AKK + a] - E[(size_t)m * AKK + a];
    p += df * df;
  }
  red[t] = p;
  __syncthreads();
  #pragma unroll
  for (int s = 64; s > 0; s >>= 1) {
    if (t < s) red[t] += red[t + s];
    __syncthreads();
  }
  float sq = red[0];
  float d = (sq > 0.0f) ? sqrtf(sq) : 0.0f;
  float c = (float)t * (1.0f / 127.0f);
  float x = d - c;
  bias[((size_t)n * NN + m) * BDB + t] = (half_t)expf(-127.0f * x * x);
}

// ---------------- generic 16x16-tiled WMMA GEMM (compile-time K, Nc) -------
// C[M,Nc] = A_h[M,K] @ Wpacked[K,Nc] + biasVec ; nullable f32/f16 outputs.
// Zero-init accumulators (inline SRC2=0), dual chains over K, bias at epilogue.
template <int K, int Nc>
__global__ void gemm16_kernel(const half_t* __restrict__ A, const half_t* __restrict__ Bp,
                              const float* __restrict__ bias,
                              float* oF0, float* oF1, half_t* oH0, half_t* oH1,
                              int M) {
  constexpr int nJ = Nc >> 4;
  constexpr int nK = K >> 5;
  int lane = threadIdx.x & 31;
  int lh = lane & 15, hi = lane >> 4;
  int tiles = (M >> 4) * nJ;
  int wid = (blockIdx.x * blockDim.x + threadIdx.x) >> 5;
  int nw = (gridDim.x * blockDim.x) >> 5;
  for (int t = wid; t < tiles; t += nw) {
    int mb = t / nJ, jb = t - mb * nJ;
    v8f c0 = {}, c1 = {};
    #pragma unroll
    for (int kk = 0; kk < nK; kk += 2) {
      {
        v16h a = load_a_frag(A, mb * 16, K, kk * 32, lane);
        v16h b = load_b_packed(Bp, kk * nJ + jb, lane);
        c0 = __builtin_amdgcn_wmma_f32_16x16x32_f16(false, a, false, b, (short)0, c0, false, false);
      }
      if (kk + 1 < nK) {
        v16h a = load_a_frag(A, mb * 16, K, (kk + 1) * 32, lane);
        v16h b = load_b_packed(Bp, (kk + 1) * nJ + jb, lane);
        c1 = __builtin_amdgcn_wmma_f32_16x16x32_f16(false, a, false, b, (short)0, c1, false, false);
      }
    }
    float bv = bias[jb * 16 + lh];
    #pragma unroll
    for (int r = 0; r < 8; ++r) {
      int mrow = mb * 16 + hi * 8 + r, col = jb * 16 + lh;
      size_t idx = (size_t)mrow * Nc + col;
      float v = c0[r] + c1[r] + bv;
      if (oF0) oF0[idx] = v;
      if (oF1) oF1[idx] = v;
      if (oH0) oH0[idx] = (half_t)v;
      if (oH1) oH1[idx] = (half_t)v;
    }
  }
}

// ---------------- THE fused bias-MLP kernel (dominant cost) ----------------
// Per 64-row block of bias [65536,128] (4 row-tiles of 16):
//   b[64,512] = biasBlk @ bpW + bpB      (WMMA, staged in dynamic LDS f16)
//   diffs[row,h] = ||b[row, h*128:+128]||
//   biasBlk    = mish(b @ bowW + bowB)   (WMMA, written back in place)
// 4 waves split the 32 (GEMM1) / 8 (GEMM2) output column-tiles; each B
// fragment feeds 4 independent row-block accumulator chains.
__global__ __launch_bounds__(128) void fused_bias_kernel(
    half_t* __restrict__ bias, const half_t* __restrict__ W1p,
    const float* __restrict__ bpB, const half_t* __restrict__ W2p,
    const float* __restrict__ bowB, float* __restrict__ diffs) {
  extern __shared__ __align__(16) half_t btile[];   // 64 * LDSLD halves
  int lane = threadIdx.x & 31;
  int wq = threadIdx.x >> 5;            // 4 waves per workgroup
  int lh = lane & 15, hi = lane >> 4;
  int rowbase = blockIdx.x * ROWS_PER_WG;

  // A fragments of the 64x128 bias block (registers, reused across all j).
  v16h a1[4][4];
  #pragma unroll
  for (int rb = 0; rb < 4; ++rb)
    #pragma unroll
    for (int kk = 0; kk < 4; ++kk)
      a1[rb][kk] = load_a_frag(bias, rowbase + rb * 16, BDB, kk * 32, lane);

  // GEMM1: each wave produces 8 of the 32 column-tiles for all 4 row-blocks.
  #pragma unroll
  for (int jj = 0; jj < 8; ++jj) {
    int j = wq * 8 + jj;
    v16h b1[4];
    #pragma unroll
    for (int kk = 0; kk < 4; ++kk)
      b1[kk] = load_b_packed(W1p, kk * 32 + j, lane);
    v8f c[4] = {};
    #pragma unroll
    for (int kk = 0; kk < 4; ++kk)
      #pragma unroll
      for (int rb = 0; rb < 4; ++rb)
        c[rb] = __builtin_amdgcn_wmma_f32_16x16x32_f16(false, a1[rb][kk], false, b1[kk], (short)0, c[rb], false, false);
    float bv = bpB[j * 16 + lh];
    #pragma unroll
    for (int rb = 0; rb < 4; ++rb)
      #pragma unroll
      for (int r = 0; r < 8; ++r)
        btile[(rb * 16 + hi * 8 + r) * LDSLD + j * 16 + lh] = (half_t)(c[rb][r] + bv);
  }
  __syncthreads();

  // per-(row, head) L2 norms of b: 256 sums, 2 per thread; b128 LDS reads.
  #pragma unroll
  for (int i = 0; i < 2; ++i) {
    int p = threadIdx.x + 128 * i;
    int row = p >> 2, head = p & 3;
    const u32x4* pp = (const u32x4*)&btile[row * LDSLD + head * 128];
    float s = 0.0f;
    #pragma unroll
    for (int c8 = 0; c8 < 16; ++c8) {
      union { u32x4 q; half_t h[8]; } u;
      u.q = pp[c8];
      #pragma unroll
      for (int e = 0; e < 8; ++e) { float v = (float)u.h[e]; s += v * v; }
    }
    diffs[(size_t)(rowbase + row) * HH + head] = sqrtf(s);
  }

  // GEMM2: each wave produces 2 of the 8 column-tiles for all 4 row-blocks.
  // Depth-1 software prefetch of next K-step's B (global) and A (LDS) frags.
  #pragma unroll
  for (int jj = 0; jj < 2; ++jj) {
    int j2 = wq * 2 + jj;
    v8f c[4] = {};
    v16h bcur = load_b_packed(W2p, 0 * 8 + j2, lane);
    v16h acur[4];
    #pragma unroll
    for (int rb = 0; rb < 4; ++rb)
      acur[rb] = load_a_frag(btile, rb * 16, LDSLD, 0, lane);
    #pragma unroll
    for (int kk = 0; kk < 16; ++kk) {
      int kn = (kk + 1) & 15;                 // wrapped: last prefetch unused
      v16h bnext = load_b_packed(W2p, kn * 8 + j2, lane);
      v16h anext[4];
      #pragma unroll
      for (int rb = 0; rb < 4; ++rb)
        anext[rb] = load_a_frag(btile, rb * 16, LDSLD, kn * 32, lane);
      #pragma unroll
      for (int rb = 0; rb < 4; ++rb)
        c[rb] = __builtin_amdgcn_wmma_f32_16x16x32_f16(false, acur[rb], false, bcur, (short)0, c[rb], false, false);
      bcur = bnext;
      #pragma unroll
      for (int rb = 0; rb < 4; ++rb) acur[rb] = anext[rb];
    }
    float bv = bowB[j2 * 16 + lh];
    #pragma unroll
    for (int rb = 0; rb < 4; ++rb)
      #pragma unroll
      for (int r = 0; r < 8; ++r) {
        float y = mishf(c[rb][r] + bv);
        bias[(size_t)(rowbase + rb * 16 + hi * 8 + r) * BDB + j2 * 16 + lh] = (half_t)y;
      }
  }
}

// ---------------- attention logits (WMMA) ----------------------------------
// logits[h,n,m] = (pq[n,h,:] . pk[m,h,:]) / sqrt(128) + diffs[n,m,h]
__global__ void logits_kernel(const half_t* __restrict__ pq, const half_t* __restrict__ pk,
                              const float* __restrict__ diffs, float* __restrict__ logits) {
  int lane = threadIdx.x & 31;
  int lh = lane & 15, hi = lane >> 4;
  int wid = (blockIdx.x * blockDim.x + threadIdx.x) >> 5;
  int nw = (gridDim.x * blockDim.x) >> 5;
  for (int t = wid; t < HH * 16 * 16; t += nw) {
    int h = t >> 8, rem = t & 255;
    int nb = rem >> 4, mb = rem & 15;
    v8f c0 = {}, c1 = {};
    #pragma unroll
    for (int kk = 0; kk < 4; kk += 2) {
      {
        v16h a = load_a_frag(pq, nb * 16, HDH, h * DHD + kk * 32, lane);
        v16h b = load_bT_rows(pk, mb * 16, HDH, h * DHD + kk * 32, lane);
        c0 = __builtin_amdgcn_wmma_f32_16x16x32_f16(false, a, false, b, (short)0, c0, false, false);
      }
      {
        v16h a = load_a_frag(pq, nb * 16, HDH, h * DHD + (kk + 1) * 32, lane);
        v16h b = load_bT_rows(pk, mb * 16, HDH, h * DHD + (kk + 1) * 32, lane);
        c1 = __builtin_amdgcn_wmma_f32_16x16x32_f16(false, a, false, b, (short)0, c1, false, false);
      }
    }
    #pragma unroll
    for (int r = 0; r < 8; ++r) {
      int n = nb * 16 + hi * 8 + r, m = mb * 16 + lh;
      logits[((size_t)h * NN + n) * NN + m] =
          (c0[r] + c1[r]) * 0.08838834764831845f + diffs[((size_t)n * NN + m) * HH + h];
    }
  }
}

// ---------------- softmax with analytic [1-I | I] mask ---------------------
// denom = sum_m e[m] ;  num = sum_m e[m]*nt[m] + e[n]*(pred[n]-nt[n])
__global__ void attn_kernel(const float* __restrict__ logits, const float* __restrict__ nt,
                            const float* __restrict__ pred, float* __restrict__ vals) {
  int lane = threadIdx.x & 31;
  int rr = (blockIdx.x * blockDim.x + threadIdx.x) >> 5;   // (h*256+n), 0..1023
  if (rr >= HH * NN) return;
  int n = rr & 255;
  const float* row = logits + (size_t)rr * NN;
  float mx = -3.0e38f;
  #pragma unroll
  for (int i = 0; i < 8; ++i) mx = fmaxf(mx, row[lane + 32 * i]);
  mx = wred_max(mx);
  float s = 0.0f, ws = 0.0f;
  #pragma unroll
  for (int i = 0; i < 8; ++i) {
    int m = lane + 32 * i;
    float e = expf(row[m] - mx);
    s += e;
    ws += e * nt[m];
  }
  s = wred_sum(s);
  ws = wred_sum(ws);
  if (lane == 0) {
    float en = expf(row[n] - mx);
    vals[rr] = (ws + en * (pred[n] - nt[n])) / s;
  }
}

__global__ void pred_update_kernel(const float* __restrict__ vals, float* __restrict__ pred) {
  int n = threadIdx.x;
  pred[n] = 0.25f * (vals[n] + vals[NN + n] + vals[2 * NN + n] + vals[3 * NN + n]);
}

// ---------------- residual + mish + LayerNorm ------------------------------
__global__ void ln_update_kernel(float* __restrict__ q, half_t* __restrict__ qh,
                                 const float* __restrict__ u, const float* __restrict__ g,
                                 const float* __restrict__ b) {
  int lane = threadIdx.x & 31;
  int rr = (blockIdx.x * blockDim.x + threadIdx.x) >> 5;
  if (rr >= NN) return;
  int c0 = lane, c1 = lane + 32;
  size_t base = (size_t)rr * DD;
  float x0 = q[base + c0] + mishf(u[base + c0]);
  float x1 = q[base + c1] + mishf(u[base + c1]);
  float mean = wred_sum(x0 + x1) * (1.0f / 64.0f);
  float d0 = x0 - mean, d1 = x1 - mean;
  float var = wred_sum(d0 * d0 + d1 * d1) * (1.0f / 64.0f);
  float rs = rsqrtf(var + 1e-5f);
  float y0 = d0 * rs * g[c0] + b[c0];
  float y1 = d1 * rs * g[c1] + b[c1];
  q[base + c0] = y0;  q[base + c1] = y1;
  qh[base + c0] = (half_t)y0;  qh[base + c1] = (half_t)y1;
}

__global__ void copy_out_kernel(const float* __restrict__ pred, float* __restrict__ out) {
  out[threadIdx.x] = pred[threadIdx.x];
}

// ---------------------------------------------------------------------------

extern "C" void kernel_launch(void* const* d_in, const int* in_sizes, int n_in,
                              void* d_out, int out_size, void* d_ws, size_t ws_size,
                              hipStream_t stream) {
  (void)in_sizes; (void)n_in; (void)out_size; (void)ws_size;
  const float* X     = (const float*)d_in[0];
  const float* E     = (const float*)d_in[1];
  const float* nt    = (const float*)d_in[2];
  const float* embW  = (const float*)d_in[3];
  const float* embB  = (const float*)d_in[4];
  const float* boutW = (const float*)d_in[5];
  const float* boutB = (const float*)d_in[6];
  const float* qpW   = (const float*)d_in[7];
  const float* qpB   = (const float*)d_in[8];
  const float* kpW   = (const float*)d_in[9];
  const float* kpB   = (const float*)d_in[10];
  const float* qoW   = (const float*)d_in[11];
  const float* qoB   = (const float*)d_in[12];
  const float* koW   = (const float*)d_in[13];
  const float* koB   = (const float*)d_in[14];
  const float* bpW   = (const float*)d_in[15];
  const float* bpB   = (const float*)d_in[16];
  const float* bowW  = (const float*)d_in[17];
  const float* bowB  = (const float*)d_in[18];
  const float* qlnG  = (const float*)d_in[19];
  const float* qlnB  = (const float*)d_in[20];
  const float* klnG  = (const float*)d_in[21];
  const float* klnB  = (const float*)d_in[22];

  char* ws = (char*)d_ws;
  size_t off = 0;
  auto alloc = [&](size_t bytes) -> char* {
    char* p = ws + off;
    off += (bytes + 255) & ~(size_t)255;
    return p;
  };

  half_t* bias   = (half_t*)alloc((size_t)RROWS * BDB * 2);   // 16 MB
  float*  diffs  = (float*) alloc((size_t)RROWS * HH * 4);    // 1 MB
  float*  logits = (float*) alloc((size_t)HH * NN * NN * 4);  // 1 MB
  half_t* pq     = (half_t*)alloc((size_t)NN * HDH * 2);
  half_t* pk     = (half_t*)alloc((size_t)NN * HDH * 2);
  float*  qf     = (float*) alloc((size_t)NN * DD * 4);
  float*  kf     = (float*) alloc((size_t)NN * DD * 4);
  half_t* qh     = (half_t*)alloc((size_t)NN * DD * 2);
  half_t* kh     = (half_t*)alloc((size_t)NN * DD * 2);
  float*  uq     = (float*) alloc((size_t)NN * DD * 4);
  float*  uk     = (float*) alloc((size_t)NN * DD * 4);
  float*  pred   = (float*) alloc((size_t)NN * 4);
  float*  vals   = (float*) alloc((size_t)HH * NN * 4);
  half_t* Xh     = (half_t*)alloc((size_t)NN * FBF * 2);
  half_t* embWp  = (half_t*)alloc((size_t)FBF * DD * 2);
  half_t* W1p    = (half_t*)alloc((size_t)LLAY * BDB * HBD * 2);
  half_t* W2p    = (half_t*)alloc((size_t)LLAY * HBD * BDB * 2);
  half_t* qpWp   = (half_t*)alloc((size_t)LLAY * DD * HDH * 2);
  half_t* kpWp   = (half_t*)alloc((size_t)LLAY * DD * HDH * 2);
  half_t* qoWp   = (half_t*)alloc((size_t)LLAY * HDH * DD * 2);
  half_t* koWp   = (half_t*)alloc((size_t)LLAY * HDH * DD * 2);

  // ---- one-time conversion / weight packing ----
  f32_to_f16_kernel<<<(NN * FBF + 255) / 256, 256, 0, stream>>>(X, Xh, NN * FBF);
  pack_b_kernel<<<(FBF * DD + 255) / 256, 256, 0, stream>>>(embW, embWp, FBF, DD);
  for (int l = 0; l < LLAY; ++l) {
    pack_b_kernel<<<(BDB * HBD + 255) / 256, 256, 0, stream>>>(
        bpW + (size_t)l * BDB * HBD, W1p + (size_t)l * BDB * HBD, BDB, HBD);
    pack_b_kernel<<<(HBD * BDB + 255) / 256, 256, 0, stream>>>(
        bowW + (size_t)l * HBD * BDB, W2p + (size_t)l * HBD * BDB, HBD, BDB);
    pack_b_kernel<<<(DD * HDH + 255) / 256, 256, 0, stream>>>(
        qpW + (size_t)l * DD * HDH, qpWp + (size_t)l * DD * HDH, DD, HDH);
    pack_b_kernel<<<(DD * HDH + 255) / 256, 256, 0, stream>>>(
        kpW + (size_t)l * DD * HDH, kpWp + (size_t)l * DD * HDH, DD, HDH);
    pack_perm_kernel<<<(HDH * DD + 255) / 256, 256, 0, stream>>>(
        qoW + (size_t)l * HDH * DD, qoWp + (size_t)l * HDH * DD, HDH, DD);
    pack_perm_kernel<<<(HDH * DD + 255) / 256, 256, 0, stream>>>(
        koW + (size_t)l * HDH * DD, koWp + (size_t)l * HDH * DD, HDH, DD);
  }

  // ---- initial state: node embedding, predictions, RBF bias ----
  gemm16_kernel<FBF, DD><<<8, 256, 0, stream>>>(Xh, embWp, embB, qf, kf, qh, kh, NN);
  pred0_kernel<<<1, 256, 0, stream>>>(X, boutW, boutB, pred);
  rbf_kernel<<<NN * NN, 128, 0, stream>>>(E, bias);

  const size_t fusedLds = (size_t)ROWS_PER_WG * LDSLD * sizeof(half_t);  // 66,560 B

  // ---- layers ----
  for (int l = 0; l < LLAY; ++l) {
    gemm16_kernel<DD, HDH><<<64, 256, 0, stream>>>(qh, qpWp + (size_t)l * DD * HDH,
                                                   qpB + (size_t)l * HDH,
                                                   nullptr, nullptr, pq, nullptr, NN);
    gemm16_kernel<DD, HDH><<<64, 256, 0, stream>>>(kh, kpWp + (size_t)l * DD * HDH,
                                                   kpB + (size_t)l * HDH,
                                                   nullptr, nullptr, pk, nullptr, NN);
    fused_bias_kernel<<<RROWS / ROWS_PER_WG, 128, fusedLds, stream>>>(
        bias, W1p + (size_t)l * BDB * HBD, bpB + (size_t)l * HBD,
        W2p + (size_t)l * HBD * BDB, bowB + (size_t)l * BDB, diffs);
    logits_kernel<<<128, 256, 0, stream>>>(pq, pk, diffs, logits);
    attn_kernel<<<128, 256, 0, stream>>>(logits, nt, pred, vals);
    pred_update_kernel<<<1, 256, 0, stream>>>(vals, pred);
    gemm16_kernel<HDH, DD><<<8, 256, 0, stream>>>(pq, qoWp + (size_t)l * HDH * DD,
                                                  qoB + (size_t)l * DD,
                                                  uq, nullptr, nullptr, nullptr, NN);
    gemm16_kernel<HDH, DD><<<8, 256, 0, stream>>>(pk, koWp + (size_t)l * HDH * DD,
                                                  koB + (size_t)l * DD,
                                                  uk, nullptr, nullptr, nullptr, NN);
    ln_update_kernel<<<32, 256, 0, stream>>>(qf, qh, uq, qlnG + (size_t)l * DD,
                                             qlnB + (size_t)l * DD);
    ln_update_kernel<<<32, 256, 0, stream>>>(kf, kh, uk, klnG + (size_t)l * DD,
                                             klnB + (size_t)l * DD);
  }

  copy_out_kernel<<<1, 256, 0, stream>>>(pred, (float*)d_out);
}
